// SpaceTimeLocalSelfAttention_45913200394593
// MI455X (gfx1250) — matrix operand order repaired and checked by reading, not compiled, not run on hardware
//
#include <hip/hip_runtime.h>
#include <cstdint>

typedef __bf16 bh;
typedef bh    v16bf __attribute__((ext_vector_type(16)));
typedef float v8f   __attribute__((ext_vector_type(8)));

#define KT  4
#define KS  8
#define B_  4
#define T_  8
#define H_  32
#define W_  32
#define C_  512
#define NH  8
#define DM  64
#define NB  4            // H_/KS
#define NW  (B_*NB*NB)   // 64 windows
#define S_  (KT*KS*KS)   // 256 tokens per window

// ---------------- helpers ----------------

__device__ __forceinline__ unsigned short f2bf(float f) {
  unsigned u = __builtin_bit_cast(unsigned, f);
  u += 0x7FFFu + ((u >> 16) & 1u);     // round-to-nearest-even
  return (unsigned short)(u >> 16);
}

struct F32B { uint4 lo, hi; };         // 32 bytes = one v16bf fragment

// A-fragment 16x32 bf16 (M x K), row-major source with leading dim ld.
// ISA layout: lane row = lane&15; lanes 0-15 hold K chunks [g*8, g*8+8) and
// [16+g*8, 16+g*8+8) -> two 16B loads, 32B apart.
__device__ __forceinline__ v16bf frag_a(const unsigned short* base, int row0,
                                        int ld, int kc, int lane) {
  const unsigned short* p = base + (row0 + (lane & 15)) * ld + kc + ((lane >> 4) << 3);
  F32B f;
  f.lo = *(const uint4*)p;
  f.hi = *(const uint4*)(p + 16);
  return __builtin_bit_cast(v16bf, f);
}

// B-fragment 32x16 bf16 (K x N) sourced from a TRANSPOSED store Bt[n][k]
// (leading dim ld): lane col = lane&15; K chunks [g*8, g*8+8) and [16+g*8, 16+g*8+8) -> two 16B loads, 32B apart.
__device__ __forceinline__ v16bf frag_b(const unsigned short* base, int n0,
                                        int ld, int kc, int lane) {
  const unsigned short* p = base + (n0 + (lane & 15)) * ld + kc + ((lane >> 4) << 3);
  F32B f;
  f.lo = *(const uint4*)p;
  f.hi = *(const uint4*)(p + 16);
  return __builtin_bit_cast(v16bf, f);
}

__device__ __forceinline__ v8f wmma_bf(v16bf a, v16bf b, v8f c) {
  return __builtin_amdgcn_wmma_f32_16x16x32_bf16(false, a, false, b, (short)0, c,
                                                 false, false);
}

// ---------------- kernel 0: one-time weight transpose+convert ----------------
// wt[sel][h][d][c] (bf16)  <-  W{q,k,v}[h][c][d] (f32)
__global__ __launch_bounds__(256) void wconv_kernel(const float* __restrict__ Wq,
                                                    const float* __restrict__ Wk,
                                                    const float* __restrict__ Wv,
                                                    unsigned short* __restrict__ wt) {
  int idx = blockIdx.x * 256 + threadIdx.x;     // < 3*8*64*512
  const float* Ws[3] = {Wq, Wk, Wv};
  int c = idx & 511, d = (idx >> 9) & 63, h = (idx >> 15) & 7, sel = idx >> 18;
  wt[idx] = f2bf(Ws[sel][(h * C_ + c) * DM + d]);
}

// ---------------- kernel 1: gather window tokens -> xb bf16 ----------------
__global__ __launch_bounds__(256) void gather_kernel(const float* __restrict__ x,
                                                     const float* __restrict__ out,
                                                     unsigned short* __restrict__ xb,
                                                     int f0, int first) {
  int wn = blockIdx.x;                  // window 0..63
  int tid = threadIdx.x;
  int s = blockIdx.y * 32 + (tid >> 3); // token 0..255
  int cbase = (tid & 7) * 64;
  int bat = wn >> 4, by = (wn >> 2) & 3, bx = wn & 3;
  int tt = s >> 6, i = (s >> 3) & 7, j = s & 7;
  int f = f0 + tt;
  const float* src = (first || tt == KT - 1) ? x : out;
  const float* p = src + ((((bat * T_ + f) * H_ + (by * KS + i)) * W_) + (bx * KS + j)) * C_ + cbase;
  unsigned short* q = xb + (wn * S_ + s) * C_ + cbase;
#pragma unroll
  for (int cc = 0; cc < 64; cc += 4) {
    float4 v = *(const float4*)(p + cc);
    uint2 w;
    w.x = (unsigned)f2bf(v.x) | ((unsigned)f2bf(v.y) << 16);
    w.y = (unsigned)f2bf(v.z) | ((unsigned)f2bf(v.w) << 16);
    *(uint2*)(q + cc) = w;
  }
}

// ---------------- kernel 2: QKV projection GEMM (WMMA) ----------------
// grid (64 windows, 8 heads, 3 sel); block 256 = 8 waves; wave tile 32x64.
__global__ __launch_bounds__(256) void proj_kernel(const unsigned short* __restrict__ xb,
                                                   const unsigned short* __restrict__ wt,
                                                   const float* __restrict__ bq,
                                                   const float* __restrict__ bk,
                                                   const float* __restrict__ bv,
                                                   unsigned short* __restrict__ qws,
                                                   unsigned short* __restrict__ kws,
                                                   unsigned short* __restrict__ vtws) {
  int wn = blockIdx.x, h = blockIdx.y, sel = blockIdx.z;
  int tid = threadIdx.x, lane = tid & 31, wv = tid >> 5;
  int R = wv * 32;
  const unsigned short* A  = xb + wn * S_ * C_;                 // [256][512]
  const unsigned short* Bt = wt + (size_t)((sel * NH + h) * DM) * C_;  // [64][512]

  v8f acc[2][4];
#pragma unroll
  for (int mi = 0; mi < 2; ++mi)
#pragma unroll
    for (int nj = 0; nj < 4; ++nj)
      acc[mi][nj] = (v8f){0.f,0.f,0.f,0.f,0.f,0.f,0.f,0.f};

  for (int kc = 0; kc < C_; kc += 32) {
    v16bf a0 = frag_a(A, R,      C_, kc, lane);
    v16bf a1 = frag_a(A, R + 16, C_, kc, lane);
#pragma unroll
    for (int nj = 0; nj < 4; ++nj) {
      v16bf b = frag_b(Bt, nj * 16, C_, kc, lane);
      acc[0][nj] = wmma_bf(a0, b, acc[0][nj]);
      acc[1][nj] = wmma_bf(a1, b, acc[1][nj]);
    }
  }

  const float* bias = (sel == 0) ? bq : (sel == 1) ? bk : bv;
  int g = lane >> 4, n16 = lane & 15;
  if (sel < 2) {
    unsigned short* dst = (sel == 0 ? qws : kws) + (size_t)(wn * NH + h) * S_ * DM;
#pragma unroll
    for (int mi = 0; mi < 2; ++mi)
#pragma unroll
      for (int nj = 0; nj < 4; ++nj) {
        float bia = bias[h * DM + nj * 16 + n16];
#pragma unroll
        for (int r = 0; r < 8; ++r) {
          int s = R + mi * 16 + g * 8 + r;
          dst[s * DM + nj * 16 + n16] = f2bf(acc[mi][nj][r] + bia);
        }
      }
  } else {  // v: store transposed [d][s] for contiguous B-fragments
    unsigned short* dst = vtws + (size_t)(wn * NH + h) * DM * S_;
#pragma unroll
    for (int mi = 0; mi < 2; ++mi)
#pragma unroll
      for (int nj = 0; nj < 4; ++nj) {
        float bia = bias[h * DM + nj * 16 + n16];
#pragma unroll
        for (int r = 0; r < 8; ++r) {
          int s = R + mi * 16 + g * 8 + r;
          dst[(nj * 16 + n16) * S_ + s] = f2bf(acc[mi][nj][r] + bia);
        }
      }
  }
}

// ---------------- kernel 3: attention (single pass, WMMA) ----------------
// grid (64 windows, 8 heads); block 256 = 8 waves; wave owns 32 query rows.
__global__ __launch_bounds__(256) void attn_kernel(const unsigned short* __restrict__ qws,
                                                   const unsigned short* __restrict__ kws,
                                                   const unsigned short* __restrict__ vtws,
                                                   const float* __restrict__ x,
                                                   float* __restrict__ out,
                                                   int f0, int first) {
  __shared__ unsigned short Pw[8][32][40];   // per-wave exp(S) chunk, padded ld=40
  int wn = blockIdx.x, h = blockIdx.y;
  int tid = threadIdx.x, lane = tid & 31, wv = tid >> 5;
  int R = wv * 32, g = lane >> 4, n16 = lane & 15;

  const unsigned short* Q  = qws  + (size_t)(wn * NH + h) * S_ * DM;
  const unsigned short* K  = kws  + (size_t)(wn * NH + h) * S_ * DM;
  const unsigned short* Vt = vtws + (size_t)(wn * NH + h) * DM * S_;
  unsigned short* P = &Pw[wv][0][0];

  v16bf qa[2][2];
#pragma unroll
  for (int kk = 0; kk < 2; ++kk) {
    qa[0][kk] = frag_a(Q, R,      DM, kk * 32, lane);
    qa[1][kk] = frag_a(Q, R + 16, DM, kk * 32, lane);
  }

  v8f o[2][4];
#pragma unroll
  for (int mi = 0; mi < 2; ++mi)
#pragma unroll
    for (int nj = 0; nj < 4; ++nj)
      o[mi][nj] = (v8f){0.f,0.f,0.f,0.f,0.f,0.f,0.f,0.f};
  float psum[16];
#pragma unroll
  for (int u = 0; u < 16; ++u) psum[u] = 0.f;

  for (int tb = 0; tb < 8; ++tb) {          // key/value chunks of 32 tokens
    v8f sc[2][2];
#pragma unroll
    for (int mi = 0; mi < 2; ++mi)
#pragma unroll
      for (int tj = 0; tj < 2; ++tj)
        sc[mi][tj] = (v8f){0.f,0.f,0.f,0.f,0.f,0.f,0.f,0.f};
#pragma unroll
    for (int kk = 0; kk < 2; ++kk) {
#pragma unroll
      for (int tj = 0; tj < 2; ++tj) {
        v16bf b = frag_b(K, tb * 32 + tj * 16, DM, kk * 32, lane);
        sc[0][tj] = wmma_bf(qa[0][kk], b, sc[0][tj]);
        sc[1][tj] = wmma_bf(qa[1][kk], b, sc[1][tj]);
      }
    }
    // exp, accumulate row sums, re-layout into A-fragment form via LDS
#pragma unroll
    for (int mi = 0; mi < 2; ++mi)
#pragma unroll
      for (int tj = 0; tj < 2; ++tj)
#pragma unroll
        for (int r = 0; r < 8; ++r) {
          float e = __expf(sc[mi][tj][r] * 0.125f);   // scale = 1/sqrt(64)
          psum[mi * 8 + r] += e;
          P[(mi * 16 + g * 8 + r) * 40 + tj * 16 + n16] = f2bf(e);
        }
    v16bf pa0 = frag_a(P, 0,  40, 0, lane);
    v16bf pa1 = frag_a(P, 16, 40, 0, lane);
#pragma unroll
    for (int nj = 0; nj < 4; ++nj) {
      v16bf b = frag_b(Vt, nj * 16, S_, tb * 32, lane);
      o[0][nj] = wmma_bf(pa0, b, o[0][nj]);
      o[1][nj] = wmma_bf(pa1, b, o[1][nj]);
    }
  }

  // softmax denominators: reduce across the 16 lanes of each half-group
#pragma unroll
  for (int u = 0; u < 16; ++u) {
    float v = psum[u];
    v += __shfl_xor(v, 1, 32);
    v += __shfl_xor(v, 2, 32);
    v += __shfl_xor(v, 4, 32);
    v += __shfl_xor(v, 8, 32);
    psum[u] = 1.0f / v;
  }

  int bat = wn >> 4, by = (wn >> 2) & 3, bx = wn & 3;
#pragma unroll
  for (int mi = 0; mi < 2; ++mi) {
    int sbase = R + mi * 16 + g * 8;        // 8-aligned: r == j within row
    int tt = sbase >> 6, i = (sbase >> 3) & 7;
    int f = f0 + tt;
    const float* src = (first || tt == KT - 1) ? x : out;
    int rowoff = ((((bat * T_ + f) * H_ + (by * KS + i)) * W_) + bx * KS) * C_;
#pragma unroll
    for (int nj = 0; nj < 4; ++nj) {
      int c = h * DM + nj * 16 + n16;
#pragma unroll
      for (int r = 0; r < 8; ++r) {
        int a = rowoff + r * C_ + c;
        out[a] = o[mi][nj][r] * psum[mi * 8 + r] + src[a];   // normalize + residual
      }
    }
  }
}

// ---------------- launcher ----------------

extern "C" void kernel_launch(void* const* d_in, const int* in_sizes, int n_in,
                              void* d_out, int out_size, void* d_ws, size_t ws_size,
                              hipStream_t stream) {
  const float* x  = (const float*)d_in[0];
  const float* Wq = (const float*)d_in[1];
  const float* Wk = (const float*)d_in[2];
  const float* Wv = (const float*)d_in[3];
  const float* bq = (const float*)d_in[4];
  const float* bk = (const float*)d_in[5];
  const float* bv = (const float*)d_in[6];
  float* out = (float*)d_out;

  char* ws = (char*)d_ws;
  const size_t MB = 1024 * 1024;
  unsigned short* xb   = (unsigned short*)(ws);              // 16 MB
  unsigned short* qws  = (unsigned short*)(ws + 16 * MB);    // 16 MB
  unsigned short* kws  = (unsigned short*)(ws + 32 * MB);    // 16 MB
  unsigned short* vtws = (unsigned short*)(ws + 48 * MB);    // 16 MB
  unsigned short* wt   = (unsigned short*)(ws + 64 * MB);    // 1.5 MB

  wconv_kernel<<<3 * NH * DM * C_ / 256, 256, 0, stream>>>(Wq, Wk, Wv, wt);

  for (int step = 0; step <= T_ - KT; ++step) {   // 5 sequential layer calls
    int f0 = step;
    int first = (step == 0) ? 1 : 0;
    gather_kernel<<<dim3(NW, 8), 256, 0, stream>>>(x, out, xb, f0, first);
    proj_kernel<<<dim3(NW, NH, 3), 256, 0, stream>>>(xb, wt, bq, bk, bv, qws, kws, vtws);
    attn_kernel<<<dim3(NW, NH), 256, 0, stream>>>(qws, kws, vtws, x, out, f0, first);
  }
}
